// OctreeDWConvBn_8761733284537
// MI455X (gfx1250) — compile-verified, matrix operand order checked by reading, not run on hardware
//
#include <hip/hip_runtime.h>

// Problem constants (fixed by the reference).
#define CCH 96          // channels
#define KNB 27          // 3x3x3 neighborhood
#define NB  1024        // pass-1 blocks == partial-sum rows (must be mult. of 32)
#define RPB 4           // rows per block per sweep
#define BN_EPS 1e-5f

typedef _Float16 f16_t;
typedef f16_t v16h __attribute__((ext_vector_type(16)));
typedef float v8f  __attribute__((ext_vector_type(8)));

// ---------------------------------------------------------------------------
// Pass 1: depthwise octree conv, write conv result to d_out, and accumulate
// per-block per-channel sum / sum-of-squares partials for BatchNorm.
// Block layout: x = channel (96 -> 3 full wave32s per row), y = row-in-tile.
// Waves see a uniform y => neighbor indices are wave-uniform (broadcast from
// LDS, uniform validity branch), and the gather data[idx*96 + c] is a fully
// coalesced 128B load per wave.
// ---------------------------------------------------------------------------
__global__ __launch_bounds__(CCH * RPB)
void octdw_pass1(const float* __restrict__ data,
                 const long long* __restrict__ neigh,
                 const float* __restrict__ weight,
                 float* __restrict__ out,
                 float* __restrict__ Psum,
                 float* __restrict__ Qsum,
                 int N)
{
    __shared__ float wlds[KNB * CCH];   // 10368 B: full weight table
    __shared__ int   ilds[RPB * KNB];   // per-tile neighbor indices
    __shared__ float rlds[CCH * RPB];   // cross-row reduction scratch

    const int tx  = threadIdx.x;        // channel
    const int ty  = threadIdx.y;        // row in tile
    const int tid = ty * CCH + tx;

    for (int i = tid; i < KNB * CCH; i += CCH * RPB) wlds[i] = weight[i];

    float s = 0.0f, s2 = 0.0f;
    const int stride = NB * RPB;
    const int iters  = (N + stride - 1) / stride;

    for (int it = 0; it < iters; ++it) {
        const int n = it * stride + (int)blockIdx.x * RPB + ty;
        __syncthreads();                         // protect ilds readers
        if (tx < KNB && n < N)
            ilds[ty * KNB + tx] = (int)neigh[(long long)n * KNB + tx];
        // Prefetch next sweep's index rows into cache (global_prefetch_b8).
        const int n2 = n + stride;
        if (tx < KNB && n2 < N)
            __builtin_prefetch(&neigh[(long long)n2 * KNB + tx], 0, 0);
        __syncthreads();                         // publish ilds
        if (n < N) {
            float acc = 0.0f;
#pragma unroll
            for (int k = 0; k < KNB; ++k) {
                const int idx = ilds[ty * KNB + k];   // wave-uniform
                if (idx >= 0)
                    acc = fmaf(data[(long long)idx * CCH + tx],
                               wlds[k * CCH + tx], acc);
            }
            out[(long long)n * CCH + tx] = acc;
            s  += acc;
            s2  = fmaf(acc, acc, s2);
        }
    }

    // Reduce the RPB per-row accumulators down to one partial per channel.
    __syncthreads();
    rlds[tid] = s;
    __syncthreads();
    if (ty == 0)
        Psum[(int)blockIdx.x * CCH + tx] =
            rlds[tx] + rlds[CCH + tx] + rlds[2 * CCH + tx] + rlds[3 * CCH + tx];
    __syncthreads();
    rlds[tid] = s2;
    __syncthreads();
    if (ty == 0)
        Qsum[(int)blockIdx.x * CCH + tx] =
            rlds[tx] + rlds[CCH + tx] + rlds[2 * CCH + tx] + rlds[3 * CCH + tx];
}

// ---------------------------------------------------------------------------
// Pass 2: reduce the 1024 x 96 partial matrices with the matrix pipe.
// D = ones(16x32) x B(32x16) + C gives column sums of B for every output row.
// f16 B operand layout: each lane holds one column (col = lane & 15), lanes
// 16..31 cover the other 16 K-rows -> lane (l) loads rows chunk*32 + (l>>4)*16
// + j. Since we only need per-column sums, the exact K->slot mapping is
// irrelevant. Finalize per-channel scale/shift for BatchNorm.
// Single wave32, EXEC all ones around the WMMAs.
// ---------------------------------------------------------------------------
__global__ __launch_bounds__(32)
void octdw_pass2_wmma(const float* __restrict__ Psum,
                      const float* __restrict__ Qsum,
                      const float* __restrict__ gamma,
                      const float* __restrict__ beta,
                      float* __restrict__ scale,
                      float* __restrict__ shift,
                      int N)
{
    const int lane = threadIdx.x;           // 0..31
    v16h a;
#pragma unroll
    for (int j = 0; j < 16; ++j) a[j] = (f16_t)1.0f;   // ones A-matrix

    const float invN = 1.0f / (float)N;
    const int col = lane & 15;
    const int rb  = (lane >> 4) * 16;

    for (int g = 0; g < CCH / 16; ++g) {
        const int c = g * 16 + col;
        v8f accS = {}; v8f accQ = {};
        for (int chunk = 0; chunk < NB / 32; ++chunk) {
            const int r0 = chunk * 32 + rb;
            v16h bS, bQ;
#pragma unroll
            for (int j = 0; j < 16; ++j) {
                bS[j] = (f16_t)Psum[(r0 + j) * CCH + c];
                bQ[j] = (f16_t)Qsum[(r0 + j) * CCH + c];
            }
            accS = __builtin_amdgcn_wmma_f32_16x16x32_f16(
                       false, a, false, bS, (short)0, accS, false, false);
            accQ = __builtin_amdgcn_wmma_f32_16x16x32_f16(
                       false, a, false, bQ, (short)0, accQ, false, false);
        }
        // Every D row holds the column sums; VGPR0 of lane L is column L&15.
        if (lane < 16) {
            const float S    = accS[0];
            const float Qv   = accQ[0];
            const float mean = S * invN;
            const float var  = fmaf(-mean, mean, Qv * invN);
            const float sc   = gamma[c] * rsqrtf(var + BN_EPS);
            scale[c] = sc;
            shift[c] = fmaf(-mean, sc, beta[c]);
        }
    }
}

// ---------------------------------------------------------------------------
// Pass 3: in-place normalize d_out with per-channel scale/shift (float4,
// fully coalesced; 96 % 4 == 0 so channel groups stay vector-aligned).
// ---------------------------------------------------------------------------
__global__ __launch_bounds__(256)
void octdw_pass3(float* __restrict__ out,
                 const float* __restrict__ scale,
                 const float* __restrict__ shift,
                 int total4)
{
    __shared__ float slds[CCH], hlds[CCH];
    if (threadIdx.x < CCH) {
        slds[threadIdx.x] = scale[threadIdx.x];
        hlds[threadIdx.x] = shift[threadIdx.x];
    }
    __syncthreads();
    const int e4 = (int)blockIdx.x * 256 + threadIdx.x;
    if (e4 < total4) {
        float4* p = reinterpret_cast<float4*>(out) + e4;
        float4 v = *p;
        const int cb = (e4 % (CCH / 4)) * 4;
        v.x = fmaf(v.x, slds[cb + 0], hlds[cb + 0]);
        v.y = fmaf(v.y, slds[cb + 1], hlds[cb + 1]);
        v.z = fmaf(v.z, slds[cb + 2], hlds[cb + 2]);
        v.w = fmaf(v.w, slds[cb + 3], hlds[cb + 3]);
        *p = v;
    }
}

// ---------------------------------------------------------------------------
extern "C" void kernel_launch(void* const* d_in, const int* in_sizes, int n_in,
                              void* d_out, int out_size, void* d_ws, size_t ws_size,
                              hipStream_t stream)
{
    const float*     data   = (const float*)d_in[0];
    const long long* neigh  = (const long long*)d_in[1];   // int64 per reference
    const float*     weight = (const float*)d_in[2];
    const float*     gamma  = (const float*)d_in[3];
    const float*     beta   = (const float*)d_in[4];
    float*           out    = (float*)d_out;

    const int N = in_sizes[0] / CCH;

    // Workspace layout (floats): P[NB*96] | Q[NB*96] | scale[96] | shift[96]
    float* Psum  = (float*)d_ws;
    float* Qsum  = Psum + NB * CCH;
    float* scale = Qsum + NB * CCH;
    float* shift = scale + CCH;

    dim3 b1(CCH, RPB);
    octdw_pass1<<<NB, b1, 0, stream>>>(data, neigh, weight, out, Psum, Qsum, N);
    octdw_pass2_wmma<<<1, 32, 0, stream>>>(Psum, Qsum, gamma, beta, scale, shift, N);
    const int total4 = (N * CCH) / 4;
    octdw_pass3<<<(total4 + 255) / 256, 256, 0, stream>>>(out, scale, shift, total4);
}
